// pointNet_27127013441921
// MI455X (gfx1250) — matrix operand order repaired and checked by reading, not compiled, not run on hardware
//
#include <hip/hip_runtime.h>
#include <hip/hip_bf16.h>

// ---------------------------------------------------------------------------
// Fused PointNet for MI455X (gfx1250).  (round-1 structure, kept)
//  - One workgroup (256 thr = 8 waves) per group of 128 points.
//  - All activations live in LDS (~251 KB of the 320 KB/WGP).
//  - GEMMs via v_wmma_f32_16x16x32_bf16, f32 accumulate.
//  - f32 residual stream held in VGPRs (160 f32/lane; gfx1250 VGPR-MSB mode
//    addresses >256 VGPRs -- seen as s_set_vgpr_msb in the asm).
//  - Weights pre-transposed + converted to bf16 in d_ws (prep kernel).
// ---------------------------------------------------------------------------

typedef __attribute__((ext_vector_type(16))) __bf16      v16bf;
typedef __attribute__((ext_vector_type(8)))  float       v8f;
typedef __attribute__((ext_vector_type(4)))  unsigned    u32x4;
typedef unsigned short u16;

constexpr int NGRP = 512;
constexpr int SEGN = 128;
constexpr int EMB  = 64;
constexpr int DIM  = 320;
constexpr int HID  = 640;
constexpr int NL   = 3;
constexpr int NRES = SEGN * DIM / 256;   // 160 residual f32 per thread

constexpr int LDA = 328;
constexpr int LDB = 648;
constexpr int LDZ = 324;

constexpr int OFF_A   = 0;
constexpr int SZ_A    = SEGN * LDA * 2;
constexpr int OFF_B   = OFF_A + SZ_A;
constexpr int SZ_B    = SEGN * LDB * 2;
constexpr int OFF_MH  = OFF_B + SZ_B;
constexpr int OFF_T   = OFF_MH + DIM * 4;
constexpr int OFF_G   = OFF_T  + HID * 4;
constexpr int OFF_GP  = OFF_G  + DIM * 4;
constexpr int OFF_RED = OFF_GP + DIM * 4;
constexpr int SMEM_TOTAL = OFF_RED + 256 * 4;      // 257280 B

constexpr size_t MSZ     = (size_t)HID * DIM;
constexpr size_t OFS_PU1 = 0;
constexpr size_t OFS_PU2 = OFS_PU1 + 3 * MSZ;
constexpr size_t OFS_GU1 = OFS_PU2 + 3 * MSZ;
constexpr size_t OFS_GU2 = OFS_GU1 + 3 * MSZ;
constexpr size_t OFS_C   = OFS_GU2 + 3 * MSZ;
constexpr size_t OFS_O1  = OFS_C   + 3 * MSZ;

__device__ __forceinline__ u16 f2bf(float f) {
  unsigned u = __float_as_uint(f);
  u += 0x7FFFu + ((u >> 16) & 1u);
  return (u16)(u >> 16);
}
__device__ __forceinline__ float bf2f(u16 h) {
  return __uint_as_float(((unsigned)h) << 16);
}
__device__ __forceinline__ float lrelu(float x) { return x >= 0.f ? x : 0.01f * x; }

union FragU { u32x4 q[2]; v16bf v; };

__device__ __forceinline__ float dot_bf16_f32(const u16* __restrict__ wrow,
                                              const float* __restrict__ x, int K) {
  float s = 0.f;
  for (int k = 0; k < K; k += 8) {
    u32x4 q = *(const u32x4*)(wrow + k);
#pragma unroll
    for (int j = 0; j < 4; ++j) {
      unsigned w2 = q[j];
      s += bf2f((u16)(w2 & 0xFFFFu)) * x[k + 2 * j];
      s += bf2f((u16)(w2 >> 16))     * x[k + 2 * j + 1];
    }
  }
  return s;
}

template <int MODE>
__device__ __forceinline__ void gemm_tile(const u16* __restrict__ A, int lda,
                                          const u16* __restrict__ W, int ldw,
                                          const float* __restrict__ bias,
                                          int KT, int NT,
                                          u16* __restrict__ Obf, int ldo,
                                          float* __restrict__ Of, int ldz,
                                          int wave, int lane) {
  const int mt   = wave;
  const int n0   = lane & 15;
  const int half = lane >> 4;
  const u16* arow = A + (size_t)(mt * 16 + n0) * lda + 8 * half;
  for (int nt = 0; nt < NT; ++nt) {
    const u16* wrow = W + (size_t)(nt * 16 + n0) * ldw + 8 * half;
    if (nt + 1 < NT)
      __builtin_prefetch(W + (size_t)((nt + 1) * 16 + n0) * ldw, 0, 3);
    v8f acc = {0.f, 0.f, 0.f, 0.f, 0.f, 0.f, 0.f, 0.f};
    for (int kt = 0; kt < KT; ++kt) {
      FragU fa, fb;
      const u16* pa = arow + kt * 32;
      fa.q[0] = *(const u32x4*)(pa);
      fa.q[1] = *(const u32x4*)(pa + 16);
      const u16* pb = wrow + kt * 32;
      fb.q[0] = *(const u32x4*)(pb);
      fb.q[1] = *(const u32x4*)(pb + 16);
      acc = __builtin_amdgcn_wmma_f32_16x16x32_bf16(false, fa.v, false, fb.v,
                                                    (short)0, acc, false, false);
    }
    const int n = nt * 16 + n0;
    float bv = (MODE == 0) ? bias[n] : 0.f;
#pragma unroll
    for (int v = 0; v < 8; ++v) {
      int m = mt * 16 + 8 * half + v;
      float x = acc[v] + bv;
      if (MODE == 0) Obf[(size_t)m * ldo + n] = f2bf(lrelu(x));
      else           Of [(size_t)m * ldz + n] = x;
    }
  }
}

extern "C" __global__ __launch_bounds__(256) void pointnet_fused(
    const int* __restrict__ words, const int* __restrict__ lengths,
    const float* __restrict__ embed,
    const float* __restrict__ pu_b1, const float* __restrict__ pu_b2,
    const float* __restrict__ gu_b1, const float* __restrict__ gu_b2,
    const float* __restrict__ c_b,
    const float* __restrict__ o_b1, const float* __restrict__ o_b2,
    const float* __restrict__ o_W2, const float* __restrict__ wscal,
    const u16* __restrict__ wPu1, const u16* __restrict__ wPu2,
    const u16* __restrict__ wGu1, const u16* __restrict__ wGu2,
    const u16* __restrict__ wC,   const u16* __restrict__ wO1,
    float* __restrict__ out) {
  extern __shared__ char smem[];
  u16*   bufA = (u16*)(smem + OFF_A);
  u16*   bufB = (u16*)(smem + OFF_B);
  float* zf   = (float*)(smem + OFF_B);
  float* mh   = (float*)(smem + OFF_MH);
  float* tv   = (float*)(smem + OFF_T);
  float* gv   = (float*)(smem + OFF_G);
  float* gp   = (float*)(smem + OFF_GP);
  float* red  = (float*)(smem + OFF_RED);

  const int gid  = blockIdx.x;
  const int tid  = threadIdx.x;
  const int wave = tid >> 5;
  const int lane = tid & 31;
  const int r0   = gid * SEGN;
  const float lenf   = (float)lengths[gid];
  const float invLen = 1.0f / lenf;

  float res[NRES];

#pragma unroll
  for (int k = 0; k < NRES; ++k) {
    int i = tid + 256 * k;
    int r = i / DIM, c = i - r * DIM;
    int letter = words[(size_t)(r0 + r) * 5 + (c >> 6)];
    float v = embed[letter * EMB + (c & 63)];
    res[k] = v;
    bufA[r * LDA + c] = f2bf(v);
  }
  __syncthreads();

  for (int l = 0; l < NL; ++l) {
    gemm_tile<0>(bufA, LDA, wPu1 + l * MSZ, DIM, pu_b1 + l * HID,
                 DIM / 32, HID / 16, bufB, LDB, nullptr, 0, wave, lane);
    __syncthreads();
    gemm_tile<0>(bufB, LDB, wPu2 + l * MSZ, HID, pu_b2 + l * DIM,
                 HID / 32, DIM / 16, bufA, LDA, nullptr, 0, wave, lane);
    __syncthreads();
    for (int c = tid; c < DIM; c += 256) {
      float s = 0.f;
      for (int r = 0; r < SEGN; ++r) s += bf2f(bufA[r * LDA + c]);
      mh[c] = s * invLen;
    }
    __syncthreads();
    for (int n = tid; n < HID; n += 256)
      tv[n] = lrelu(gu_b1[l * HID + n] +
                    dot_bf16_f32(wGu1 + l * MSZ + (size_t)n * DIM, mh, DIM));
    __syncthreads();
    for (int n = tid; n < DIM; n += 256)
      gv[n] = lrelu(gu_b2[l * DIM + n] +
                    dot_bf16_f32(wGu2 + l * MSZ + (size_t)n * HID, tv, HID));
    __syncthreads();
    for (int n = tid; n < DIM; n += 256)
      gp[n] = c_b[l * DIM + n] +
              dot_bf16_f32(wC + l * MSZ + (size_t)n * HID + DIM, gv, DIM);
    gemm_tile<1>(bufA, LDA, wC + l * MSZ, HID, nullptr,
                 DIM / 32, DIM / 16, nullptr, 0, zf, LDZ, wave, lane);
    __syncthreads();
#pragma unroll
    for (int k = 0; k < NRES; ++k) {
      int i = tid + 256 * k;
      int r = i / DIM, c = i - r * DIM;
      float v = lrelu(zf[r * LDZ + c] + gp[c]) + res[k];
      res[k] = v;
      bufA[r * LDA + c] = f2bf(v);
    }
    __syncthreads();
  }

  for (int c = tid; c < DIM; c += 256) mh[c] = 0.f;
  __syncthreads();
#pragma unroll
  for (int k = 0; k < NRES; ++k) {
    int i = tid + 256 * k;
    atomicAdd(&mh[i % DIM], res[k] * invLen);   // ds_add_f32 segment mean
  }
  __syncthreads();
  for (int n = tid; n < HID; n += 256)
    tv[n] = lrelu(o_b1[n] + dot_bf16_f32(wO1 + (size_t)n * DIM, mh, DIM));
  __syncthreads();
  float part = 0.f;
  for (int n = tid; n < HID; n += 256) part += tv[n] * o_W2[n];
  red[tid] = part;
  __syncthreads();
  for (int s = 128; s > 0; s >>= 1) {
    if (tid < s) red[tid] += red[tid + s];
    __syncthreads();
  }
  if (tid == 0) {
    float y = lrelu(red[0] + o_b2[0]);
    out[gid] = y + wscal[0] * log2f(lenf);
  }
}

extern "C" __global__ void prep_transpose_bf16(const float* __restrict__ src,
                                               u16* __restrict__ dst,
                                               int rows, int cols, int batch) {
  size_t total  = (size_t)batch * rows * cols;
  size_t stride = (size_t)gridDim.x * blockDim.x;
  for (size_t i = (size_t)blockIdx.x * blockDim.x + threadIdx.x; i < total; i += stride) {
    size_t mat = i / ((size_t)rows * cols);
    size_t rem = i - mat * (size_t)rows * cols;
    int r = (int)(rem / cols);
    int c = (int)(rem - (size_t)r * cols);
    dst[mat * (size_t)rows * cols + (size_t)c * rows + r] = f2bf(src[i]);
  }
}

extern "C" void kernel_launch(void* const* d_in, const int* in_sizes, int n_in,
                              void* d_out, int out_size, void* d_ws, size_t ws_size,
                              hipStream_t stream) {
  (void)in_sizes; (void)n_in; (void)out_size; (void)ws_size;
  const int*   words   = (const int*)d_in[0];
  const int*   lengths = (const int*)d_in[1];
  const float* embed = (const float*)d_in[3];
  const float* pu_W1 = (const float*)d_in[4];
  const float* pu_b1 = (const float*)d_in[5];
  const float* pu_W2 = (const float*)d_in[6];
  const float* pu_b2 = (const float*)d_in[7];
  const float* gu_W1 = (const float*)d_in[8];
  const float* gu_b1 = (const float*)d_in[9];
  const float* gu_W2 = (const float*)d_in[10];
  const float* gu_b2 = (const float*)d_in[11];
  const float* c_W   = (const float*)d_in[12];
  const float* c_b   = (const float*)d_in[13];
  const float* o_W1  = (const float*)d_in[14];
  const float* o_b1  = (const float*)d_in[15];
  const float* o_W2  = (const float*)d_in[16];
  const float* o_b2  = (const float*)d_in[17];
  const float* wscal = (const float*)d_in[18];

  u16* ws = (u16*)d_ws;
  dim3 pg(512), pb(256);
  prep_transpose_bf16<<<pg, pb, 0, stream>>>(pu_W1, ws + OFS_PU1, DIM, HID, 3);
  prep_transpose_bf16<<<pg, pb, 0, stream>>>(pu_W2, ws + OFS_PU2, HID, DIM, 3);
  prep_transpose_bf16<<<pg, pb, 0, stream>>>(gu_W1, ws + OFS_GU1, DIM, HID, 3);
  prep_transpose_bf16<<<pg, pb, 0, stream>>>(gu_W2, ws + OFS_GU2, HID, DIM, 3);
  prep_transpose_bf16<<<pg, pb, 0, stream>>>(c_W,   ws + OFS_C,   HID, DIM, 3);
  prep_transpose_bf16<<<pg, pb, 0, stream>>>(o_W1,  ws + OFS_O1,  DIM, HID, 1);

  hipFuncSetAttribute(reinterpret_cast<const void*>(pointnet_fused),
                      hipFuncAttributeMaxDynamicSharedMemorySize, SMEM_TOTAL);
  pointnet_fused<<<dim3(NGRP), dim3(256), SMEM_TOTAL, stream>>>(
      words, lengths, embed, pu_b1, pu_b2, gu_b1, gu_b2, c_b, o_b1, o_b2,
      o_W2, wscal, ws + OFS_PU1, ws + OFS_PU2, ws + OFS_GU1, ws + OFS_GU2,
      ws + OFS_C, ws + OFS_O1, (float*)d_out);
}
